// NAAGAT_24481313587856
// MI455X (gfx1250) — matrix operand order repaired and checked by reading, not compiled, not run on hardware
//
#include <hip/hip_runtime.h>

#define N_NODES 50000
#define N_EDGES 1600000
#define E_TOT   1650000      // edges + one self loop per node
#define IN_DIM  256
#define HID     128
#define HEADS   4
#define C1      32
#define NEG_SLOPE 0.2f

typedef __attribute__((ext_vector_type(16))) __bf16          v16bf;
typedef __attribute__((ext_vector_type(8)))  float           v8f;
typedef __attribute__((ext_vector_type(8)))  unsigned short  v8u;
typedef __attribute__((ext_vector_type(16))) unsigned short  v16u;
typedef __attribute__((ext_vector_type(4)))  float           f32x4;
typedef __attribute__((ext_vector_type(4)))  unsigned int    u32x4;

__device__ __forceinline__ unsigned short f2bf(float f) {
  unsigned u = __float_as_uint(f);
  u += 0x7FFFu + ((u >> 16) & 1u);       // round-to-nearest-even
  return (unsigned short)(u >> 16);
}

__device__ __forceinline__ float lrelu(float x) { return x > 0.f ? x : NEG_SLOPE * x; }

// IEEE-order float atomic max via signed/unsigned integer atomics.
__device__ __forceinline__ void atomicMaxF(float* a, float v) {
  if (v >= 0.f) atomicMax((int*)a,          __float_as_int(v));
  else          atomicMin((unsigned int*)a, __float_as_uint(v));
}

// ---------------------------------------------------------------- init
__global__ void k0_fill(float* p, float v, int n) {
  for (int i = blockIdx.x * blockDim.x + threadIdx.x; i < n; i += gridDim.x * blockDim.x)
    p[i] = v;
}

// ------------------------------------------- scale by sigmoid(fi), ->bf16
__global__ void k1_scale_convert(const float* __restrict__ x, const float* __restrict__ fi,
                                 unsigned short* __restrict__ xw) {
  int t = blockIdx.x * blockDim.x + threadIdx.x;
  if (t >= N_NODES * IN_DIM) return;
  int k = t & (IN_DIM - 1);
  float g = 1.f / (1.f + __expf(-fi[k]));
  xw[t] = f2bf(x[t] * g);
}

// ---------------------------------------------------- W1^T -> bf16 [128][256]
__global__ void k1b_w1t(const float* __restrict__ W1, unsigned short* __restrict__ w1t) {
  int t = blockIdx.x * blockDim.x + threadIdx.x;
  if (t >= HID * IN_DIM) return;
  int c = t >> 8, k = t & 255;
  w1t[t] = f2bf(W1[k * HID + c]);
}

// ------------------------------------------- GEMM1: xp1 = xw @ W1  (WMMA bf16)
__global__ __launch_bounds__(128) void gat_gemm1_wmma(const unsigned short* __restrict__ A,
                                                      const unsigned short* __restrict__ Bt,
                                                      float* __restrict__ C) {
  __shared__ unsigned short As[64][40];    // 64 rows x 32 K, padded to 40 (80 B stride)
  __shared__ unsigned short Bs[128][40];   // 128 cols x 32 K (B^T layout)
  const int tid  = threadIdx.x;
  const int wave = tid >> 5;
  const int lane = tid & 31;
  const int m    = lane & 15;
  const int g    = lane >> 4;
  const int rowBase = blockIdx.x * 64;

  v8f acc[8] = {};

  for (int k0 = 0; k0 < IN_DIM; k0 += 32) {
    __syncthreads();
    { // stage A tile: each thread copies 32 B
      int r = tid >> 1, seg = tid & 1;
      int gr = rowBase + r; if (gr >= N_NODES) gr = N_NODES - 1;
      const u32x4* gp = (const u32x4*)(A + (size_t)gr * IN_DIM + k0 + seg * 16);
      u32x4 v0 = gp[0], v1 = gp[1];
      *(u32x4*)&As[r][seg * 16]     = v0;
      *(u32x4*)&As[r][seg * 16 + 8] = v1;
    }
    { // stage B tile: each thread copies one 64 B row of W1^T
      const u32x4* gp = (const u32x4*)(Bt + (size_t)tid * IN_DIM + k0);
      u32x4 b0 = gp[0], b1v = gp[1], b2v = gp[2], b3v = gp[3];
      *(u32x4*)&Bs[tid][0]  = b0;
      *(u32x4*)&Bs[tid][8]  = b1v;
      *(u32x4*)&Bs[tid][16] = b2v;
      *(u32x4*)&Bs[tid][24] = b3v;
    }
    __syncthreads();

    // A fragment per ISA: lane(m,g): elems 0..7 = K 8g.., elems 8..15 = K 16+8g..
    const unsigned short* ap = &As[wave * 16 + m][0];
    v8u alo = *(const v8u*)(ap + 8 * g);
    v8u ahi = *(const v8u*)(ap + 16 + 8 * g);
    v16u au = __builtin_shufflevector(alo, ahi, 0,1,2,3,4,5,6,7,8,9,10,11,12,13,14,15);
    v16bf afrag = __builtin_bit_cast(v16bf, au);

#pragma unroll
    for (int nt = 0; nt < 8; ++nt) {
      // B fragment: lane(n,g): elems 0..7 = K 8g.., elems 8..15 = K 16+8g.. at column n
      const unsigned short* bp = &Bs[nt * 16 + m][0];
      v8u blo = *(const v8u*)(bp + 8 * g);
      v8u bhi = *(const v8u*)(bp + 16 + 8 * g);
      v16u bu = __builtin_shufflevector(blo, bhi, 0,1,2,3,4,5,6,7,8,9,10,11,12,13,14,15);
      v16bf bfrag = __builtin_bit_cast(v16bf, bu);
      acc[nt] = __builtin_amdgcn_wmma_f32_16x16x32_bf16(false, afrag, false, bfrag,
                                                        (short)0, acc[nt], false, false);
    }
  }

  // C/D layout: VGPR r, lanes 0-15: M=r, N=lane; lanes 16-31: M=r+8
#pragma unroll
  for (int nt = 0; nt < 8; ++nt) {
#pragma unroll
    for (int r = 0; r < 8; ++r) {
      int row = rowBase + wave * 16 + r + 8 * g;
      if (row < N_NODES) C[(size_t)row * HID + nt * 16 + m] = acc[nt][r];
    }
  }
}

// ------------------------------------------- per-(node,head) attn logits L1
__global__ void k3_attn1(const float* __restrict__ xp1, const float* __restrict__ att_src1,
                         const float* __restrict__ att_dst1,
                         float* __restrict__ a_src1, float* __restrict__ a_dst1) {
  int t = blockIdx.x * blockDim.x + threadIdx.x;
  if (t >= N_NODES * HEADS) return;
  int n = t >> 2, h = t & 3;
  const float* row = xp1 + (size_t)n * HID + h * C1;
  const float* as  = att_src1 + h * C1;
  const float* ad  = att_dst1 + h * C1;
  float s = 0.f, d = 0.f;
#pragma unroll 8
  for (int j = 0; j < C1; ++j) { s += row[j] * as[j]; d += row[j] * ad[j]; }
  a_src1[t] = s; a_dst1[t] = d;
}

// ------------------------------------------- edge pass A: segment max (L1)
__global__ __launch_bounds__(256) void k4_max1(const long long* __restrict__ ei,
                                               const float* __restrict__ a_src1,
                                               const float* __restrict__ a_dst1,
                                               float* __restrict__ max1) {
  int t = blockIdx.x * blockDim.x + threadIdx.x;
  if (t >= E_TOT * 4) return;
  int e = t >> 2, h = t & 3;
  __builtin_prefetch(ei + e + 4096, 0, 0);
  long long s, d;
  if (e < N_EDGES) { s = ei[e]; d = ei[N_EDGES + e]; } else { s = d = (long long)(e - N_EDGES); }
  float lr = lrelu(a_src1[s * 4 + h] + a_dst1[d * 4 + h]);
  atomicMaxF(&max1[d * 4 + h], lr);
}

// ---- edge pass B: e=exp(lr-max); sum += e; out += e * xp[src]  (wave/edge)
__global__ __launch_bounds__(256) void k5_scatter1(const long long* __restrict__ ei,
                                                   const float* __restrict__ a_src1,
                                                   const float* __restrict__ a_dst1,
                                                   const float* __restrict__ max1,
                                                   const float* __restrict__ xp1,
                                                   float* __restrict__ sum1,
                                                   float* __restrict__ out1) {
  int gid  = blockIdx.x * 256 + threadIdx.x;
  int e    = gid >> 5;
  int lane = gid & 31;
  if (e >= E_TOT) return;
  long long s, d;
  if (e < N_EDGES) { s = ei[e]; d = ei[N_EDGES + e]; } else { s = d = (long long)(e - N_EDGES); }
  float my_e = 0.f;
  if (lane < HEADS) {
    float lr = lrelu(a_src1[s * 4 + lane] + a_dst1[d * 4 + lane]);
    my_e = __expf(lr - max1[d * 4 + lane]);
    atomicAdd(&sum1[d * 4 + lane], my_e);
  }
  float eh = __shfl(my_e, lane >> 3, 32);                 // head = lane/8
  f32x4 v = *(const f32x4*)(xp1 + (size_t)s * HID + lane * 4);
  float* o = out1 + (size_t)d * HID + lane * 4;
  atomicAdd(o + 0, v[0] * eh);
  atomicAdd(o + 1, v[1] * eh);
  atomicAdd(o + 2, v[2] * eh);
  atomicAdd(o + 3, v[3] * eh);
}

// ---- finalize L1 (normalize + bias + ELU), fuse GEMM2 + L2 attn logits
__global__ __launch_bounds__(256) void k6_final1(float* __restrict__ out1,
                                                 const float* __restrict__ sum1,
                                                 const float* __restrict__ b1,
                                                 const float* __restrict__ W2,
                                                 const float* __restrict__ att_src2,
                                                 const float* __restrict__ att_dst2,
                                                 float* __restrict__ xp2,
                                                 float* __restrict__ a_src2,
                                                 float* __restrict__ a_dst2) {
  int gid = blockIdx.x * 256 + threadIdx.x;
  int n = gid >> 5, lane = gid & 31;
  if (n >= N_NODES) return;
  int head = lane >> 3;
  float inv = 1.f / (sum1[n * 4 + head] + 1e-16f);
  float* rowp = out1 + (size_t)n * HID + lane * 4;
  f32x4 v = *(f32x4*)rowp;
  float p0 = 0.f, p1 = 0.f;
#pragma unroll
  for (int i = 0; i < 4; ++i) {
    int c = lane * 4 + i;
    float hv = v[i] * inv + b1[c];
    hv = hv > 0.f ? hv : (__expf(hv) - 1.f);              // ELU (alpha=1)
    v[i] = hv;
    p0 += hv * W2[c * 2 + 0];
    p1 += hv * W2[c * 2 + 1];
  }
  *(f32x4*)rowp = v;
#pragma unroll
  for (int off = 16; off > 0; off >>= 1) {
    p0 += __shfl_down(p0, off, 32);
    p1 += __shfl_down(p1, off, 32);
  }
  if (lane == 0) {
    xp2[n * 2 + 0] = p0; xp2[n * 2 + 1] = p1;
    a_src2[n] = p0 * att_src2[0] + p1 * att_src2[1];
    a_dst2[n] = p0 * att_dst2[0] + p1 * att_dst2[1];
  }
}

// ------------------------------------------- L2 edge passes (single head)
__global__ __launch_bounds__(256) void k7_max2(const long long* __restrict__ ei,
                                               const float* __restrict__ a_src2,
                                               const float* __restrict__ a_dst2,
                                               float* __restrict__ max2) {
  int e = blockIdx.x * blockDim.x + threadIdx.x;
  if (e >= E_TOT) return;
  long long s, d;
  if (e < N_EDGES) { s = ei[e]; d = ei[N_EDGES + e]; } else { s = d = (long long)(e - N_EDGES); }
  atomicMaxF(&max2[d], lrelu(a_src2[s] + a_dst2[d]));
}

__global__ __launch_bounds__(256) void k8_scatter2(const long long* __restrict__ ei,
                                                   const float* __restrict__ a_src2,
                                                   const float* __restrict__ a_dst2,
                                                   const float* __restrict__ max2,
                                                   const float* __restrict__ xp2,
                                                   float* __restrict__ sum2,
                                                   float* __restrict__ out2) {
  int e = blockIdx.x * blockDim.x + threadIdx.x;
  if (e >= E_TOT) return;
  long long s, d;
  if (e < N_EDGES) { s = ei[e]; d = ei[N_EDGES + e]; } else { s = d = (long long)(e - N_EDGES); }
  float lr = lrelu(a_src2[s] + a_dst2[d]);
  float ev = __expf(lr - max2[d]);
  atomicAdd(&sum2[d], ev);
  atomicAdd(&out2[d * 2 + 0], ev * xp2[s * 2 + 0]);
  atomicAdd(&out2[d * 2 + 1], ev * xp2[s * 2 + 1]);
}

__global__ void k9_final2(const float* __restrict__ out2, const float* __restrict__ sum2,
                          const float* __restrict__ b2, float* __restrict__ out) {
  int t = blockIdx.x * blockDim.x + threadIdx.x;
  if (t >= N_NODES * 2) return;
  int n = t >> 1, c = t & 1;
  out[t] = out2[t] / (sum2[n] + 1e-16f) + b2[c];
}

// ================================================================= launch
extern "C" void kernel_launch(void* const* d_in, const int* in_sizes, int n_in,
                              void* d_out, int out_size, void* d_ws, size_t ws_size,
                              hipStream_t stream) {
  const float*     x   = (const float*)d_in[0];
  const float*     fi  = (const float*)d_in[1];
  const float*     W1  = (const float*)d_in[2];
  const float*     as1 = (const float*)d_in[3];
  const float*     ad1 = (const float*)d_in[4];
  const float*     b1  = (const float*)d_in[5];
  const float*     W2  = (const float*)d_in[6];
  const float*     as2 = (const float*)d_in[7];
  const float*     ad2 = (const float*)d_in[8];
  const float*     b2  = (const float*)d_in[9];
  const long long* ei  = (const long long*)d_in[10];   // int64 edge_index [2, E]
  float* out = (float*)d_out;

  char* p = (char*)d_ws;
  auto carve = [&](size_t bytes) { char* r = p; p += (bytes + 255) & ~(size_t)255; return r; };
  unsigned short* xw     = (unsigned short*)carve((size_t)N_NODES * IN_DIM * 2);
  unsigned short* w1t    = (unsigned short*)carve((size_t)HID * IN_DIM * 2);
  float*          xp1    = (float*)carve((size_t)N_NODES * HID * 4);
  float*          a_src1 = (float*)carve((size_t)N_NODES * HEADS * 4);
  float*          a_dst1 = (float*)carve((size_t)N_NODES * HEADS * 4);
  float*          max1   = (float*)carve((size_t)N_NODES * HEADS * 4);
  float*          sum1   = (float*)carve((size_t)N_NODES * HEADS * 4);
  float*          out1   = (float*)carve((size_t)N_NODES * HID * 4);
  float*          xp2    = (float*)carve((size_t)N_NODES * 2 * 4);
  float*          a_src2 = (float*)carve((size_t)N_NODES * 4);
  float*          a_dst2 = (float*)carve((size_t)N_NODES * 4);
  float*          max2   = (float*)carve((size_t)N_NODES * 4);
  float*          sum2   = (float*)carve((size_t)N_NODES * 4);
  float*          out2   = (float*)carve((size_t)N_NODES * 2 * 4);

  const float NEG_INF = -__builtin_huge_valf();
  k0_fill<<<512, 256, 0, stream>>>(max1, NEG_INF, N_NODES * HEADS);
  k0_fill<<<512, 256, 0, stream>>>(sum1, 0.f, N_NODES * HEADS);
  k0_fill<<<2048, 256, 0, stream>>>(out1, 0.f, N_NODES * HID);
  k0_fill<<<256, 256, 0, stream>>>(max2, NEG_INF, N_NODES);
  k0_fill<<<256, 256, 0, stream>>>(sum2, 0.f, N_NODES);
  k0_fill<<<256, 256, 0, stream>>>(out2, 0.f, N_NODES * 2);

  k1_scale_convert<<<(N_NODES * IN_DIM + 255) / 256, 256, 0, stream>>>(x, fi, xw);
  k1b_w1t<<<(HID * IN_DIM + 255) / 256, 256, 0, stream>>>(W1, w1t);
  gat_gemm1_wmma<<<(N_NODES + 63) / 64, 128, 0, stream>>>(xw, w1t, xp1);
  k3_attn1<<<(N_NODES * HEADS + 255) / 256, 256, 0, stream>>>(xp1, as1, ad1, a_src1, a_dst1);
  k4_max1<<<(E_TOT * 4 + 255) / 256, 256, 0, stream>>>(ei, a_src1, a_dst1, max1);
  k5_scatter1<<<(E_TOT * 32 + 255) / 256, 256, 0, stream>>>(ei, a_src1, a_dst1, max1, xp1, sum1, out1);
  k6_final1<<<(N_NODES * 32 + 255) / 256, 256, 0, stream>>>(out1, sum1, b1, W2, as2, ad2,
                                                            xp2, a_src2, a_dst2);
  k7_max2<<<(E_TOT + 255) / 256, 256, 0, stream>>>(ei, a_src2, a_dst2, max2);
  k8_scatter2<<<(E_TOT + 255) / 256, 256, 0, stream>>>(ei, a_src2, a_dst2, max2, xp2, sum2, out2);
  k9_final2<<<(N_NODES * 2 + 255) / 256, 256, 0, stream>>>(out2, sum2, b2, out);

  (void)in_sizes; (void)n_in; (void)out_size; (void)ws_size;
}
